// GRULayer_23441931501722
// MI455X (gfx1250) — matrix operand order repaired and checked by reading, not compiled, not run on hardware
//
#include <hip/hip_runtime.h>

typedef __attribute__((ext_vector_type(16))) _Float16 v16h;
typedef __attribute__((ext_vector_type(8)))  _Float16 v8h;
typedef __attribute__((ext_vector_type(8)))  float    v8f;
typedef __attribute__((ext_vector_type(4)))  float    v4f;

#define BB 32
#define SS 48
#define NSEQ 307
#define DD 64
#define GG 192
#define BN (BB*NSEQ)   /* 9824 = 614 tiles of 16 */
#define WROW 72        /* padded LDS row stride in halfs (144B -> conflict-free) */

// ---- hardware tanh (CDNA5 V_TANH_F32) ----
__device__ __forceinline__ float htanh(float x) {
#if __has_builtin(__builtin_amdgcn_tanhf)
    return __builtin_amdgcn_tanhf(x);
#elif __has_builtin(__builtin_amdgcn_tanh_f32)
    return __builtin_amdgcn_tanh_f32(x);
#else
    float y;
    asm("v_tanh_f32 %0, %1" : "=v"(y) : "v"(x));
    return y;
#endif
}

__device__ __forceinline__ v8f wmma_f16(v16h a, v16h b, v8f c) {
    // D = A(16x32 f16) * B(32x16 f16) + C(16x16 f32)
    return __builtin_amdgcn_wmma_f32_16x16x32_f16(
        false, a, false, b, (short)0, c, false, false);
}

__device__ __forceinline__ v16h join16(v8h lo, v8h hi) {
    return __builtin_shufflevector(lo, hi, 0,1,2,3,4,5,6,7,8,9,10,11,12,13,14,15);
}

// B-operand fragment: 16 contiguous halfs starting at p
__device__ __forceinline__ v16h loadB(const _Float16* p) {
    const v8h* vp = (const v8h*)p;
    return join16(vp[0], vp[1]);
}

// A-operand fragment from 2 runs of 8 contiguous f32 in global memory
__device__ __forceinline__ v16h loadA_x(const float* p0, const float* p1) {
    v4f a0 = *(const v4f*)p0, a1 = *(const v4f*)(p0 + 4);
    v4f b0 = *(const v4f*)p1, b1 = *(const v4f*)(p1 + 4);
    v16h r;
    #pragma unroll
    for (int i = 0; i < 4; ++i) {
        r[i]      = (_Float16)a0[i];
        r[4 + i]  = (_Float16)a1[i];
        r[8 + i]  = (_Float16)b0[i];
        r[12 + i] = (_Float16)b1[i];
    }
    return r;
}

__global__ __launch_bounds__(128, 1)
#if __has_attribute(amdgpu_num_vgpr)
__attribute__((amdgpu_num_vgpr(256)))
#endif
void gru_fused_wmma(const float* __restrict__ x,    // [B,S,N,D]
                    const float* __restrict__ Wih,  // [192,64]
                    const float* __restrict__ Whh,  // [192,64]
                    const float* __restrict__ bih,  // [192]
                    const float* __restrict__ bhh,  // [192]
                    float* __restrict__ out)        // [S, BN, D]
{
    __shared__ __align__(16) _Float16 sW[2][GG][WROW];  // f16 weights, row-major
    __shared__ __align__(16) _Float16 sH[2][16][WROW];  // double-buffered h tile

    const int tid  = threadIdx.x;
    const int lane = tid & 31;
    const int c    = tid >> 5;      // wave id == d-slice (0..3)
    const int ln   = lane & 15;
    const int lh   = lane >> 4;     // lane-half

    // ---- one-time: weights -> LDS as f16; zero h buffer 0 ----
    for (int i = tid; i < GG * DD; i += 128) {
        int g = i >> 6, d = i & 63;
        sW[0][g][d] = (_Float16)Wih[i];
        sW[1][g][d] = (_Float16)Whh[i];
    }
    for (int i = tid; i < 16 * WROW; i += 128)
        ((_Float16*)sH[0])[i] = (_Float16)0.0f;
    __syncthreads();

    // ---- per-lane row/base setup (one 16-row tile per block) ----
    const int rowbase = blockIdx.x * 16;
    const int rr = rowbase + ln;                 // this lane's A-row (m = ln)
    const int bidx = rr / NSEQ, nidx = rr % NSEQ;
    const float* xlane = x + ((size_t)(bidx * SS) * NSEQ + nidx) * DD;

    // biases for this wave's d-slice d0 = 16c+ln
    const int d0 = 16 * c + ln;
    const float hbsr = 0.5f * (bih[d0]      + bhh[d0]);       // r (pre-halved)
    const float hbsz = 0.5f * (bih[64 + d0] + bhh[64 + d0]);  // z (pre-halved)
    const float bin_ = bih[128 + d0];                         // n (input side)
    const float bhn  = bhh[128 + d0];                         // n (hidden side)

    // ---- pin this wave's 12 weight B-fragments in VGPRs (zero loads/step) ----
    v16h wihB[3][2], whhB[3][2];
    #pragma unroll
    for (int g = 0; g < 3; ++g) {
        const int row = (g * 4 + c) * 16 + ln;   // gate g, col-tile c
        #pragma unroll
        for (int kt = 0; kt < 2; ++kt) {
            wihB[g][kt] = join16(*(const v8h*)&sW[0][row][32 * kt + 8 * lh],
                                 *(const v8h*)&sW[0][row][32 * kt + 16 + 8 * lh]);
            whhB[g][kt] = join16(*(const v8h*)&sW[1][row][32 * kt + 8 * lh],
                                 *(const v8h*)&sW[1][row][32 * kt + 16 + 8 * lh]);
        }
    }

    v8f hold = (v8f)0.0f;

    #pragma unroll 1
    for (int s = 0; s < SS; ++s) {
        const int pr = s & 1;        // read buffer; write buffer = 1-pr

        const float* xp = xlane + (size_t)s * (NSEQ * DD);
        // A fragments for x_t : k-runs are 8 contiguous f32
        v16h xa0 = loadA_x(xp +      8 * lh, xp + 16 + 8 * lh);
        v16h xa1 = loadA_x(xp + 32 + 8 * lh, xp + 48 + 8 * lh);
        if (s + 1 < SS) {
            __builtin_prefetch(xp + (size_t)(NSEQ * DD) + 8 * lh, 0, 1);
            __builtin_prefetch(xp + (size_t)(NSEQ * DD) + 32 + 8 * lh, 0, 1);
        }
        // A fragments for h_{t-1} from LDS (two 16B runs per K-tile)
        v16h ha0 = join16(*(const v8h*)&sH[pr][ln][ 0 + 8 * lh],
                          *(const v8h*)&sH[pr][ln][16 + 8 * lh]);
        v16h ha1 = join16(*(const v8h*)&sH[pr][ln][32 + 8 * lh],
                          *(const v8h*)&sH[pr][ln][48 + 8 * lh]);

        v8f hr = (v8f)0.0f, hz = (v8f)0.0f, hn  = (v8f)0.0f;
        v8f ir = (v8f)0.0f, iz = (v8f)0.0f, inq = (v8f)0.0f;
        hr  = wmma_f16(ha0, whhB[0][0], hr);
        hz  = wmma_f16(ha0, whhB[1][0], hz);
        hn  = wmma_f16(ha0, whhB[2][0], hn);
        ir  = wmma_f16(xa0, wihB[0][0], ir);
        iz  = wmma_f16(xa0, wihB[1][0], iz);
        inq = wmma_f16(xa0, wihB[2][0], inq);
        hr  = wmma_f16(ha1, whhB[0][1], hr);
        hz  = wmma_f16(ha1, whhB[1][1], hz);
        hn  = wmma_f16(ha1, whhB[2][1], hn);
        ir  = wmma_f16(xa1, wihB[0][1], ir);
        iz  = wmma_f16(xa1, wihB[1][1], iz);
        inq = wmma_f16(xa1, wihB[2][1], inq);

        // gates + state update; D-layout: element e -> (m = e+8*lh, d = 16c+ln)
        float* op = out + ((size_t)s * BN + rowbase + 8 * lh) * DD + d0;
        #pragma unroll
        for (int e = 0; e < 8; ++e) {
            // r,z: sigmoid(a) = 0.5*tanh(a/2)+0.5 (one V_TANH_F32 each)
            float r  = fmaf(0.5f, htanh(fmaf(0.5f, ir[e] + hr[e], hbsr)), 0.5f);
            float z  = fmaf(0.5f, htanh(fmaf(0.5f, iz[e] + hz[e], hbsz)), 0.5f);
            float ng = htanh(fmaf(r, hn[e] + bhn, inq[e] + bin_));
            float h  = fmaf(z, hold[e] - ng, ng);   // (1-z)*ng + z*h_old
            hold[e] = h;
            op[(size_t)e * DD] = h;
            sH[1 - pr][e + 8 * lh][d0] = (_Float16)h;
        }
        __syncthreads();   // writes(1-pr) visible before next step's reads
    }
}

extern "C" void kernel_launch(void* const* d_in, const int* in_sizes, int n_in,
                              void* d_out, int out_size, void* d_ws, size_t ws_size,
                              hipStream_t stream) {
    const float* x   = (const float*)d_in[0];
    const float* Wih = (const float*)d_in[1];
    const float* Whh = (const float*)d_in[2];
    const float* bih = (const float*)d_in[3];
    const float* bhh = (const float*)d_in[4];
    float* out = (float*)d_out;

    dim3 grid(BN / 16);   // 614 blocks; 4 waves/block (one d-slice each)
    dim3 block(128);
    hipLaunchKernelGGL(gru_fused_wmma, grid, block, 0, stream,
                       x, Wih, Whh, bih, bhh, out);
}
